// MultiHeadSelfAttention_64819646431786
// MI455X (gfx1250) — matrix operand order repaired and checked by reading, not compiled, not run on hardware
//
#include <hip/hip_runtime.h>
#include <hip/hip_bf16.h>

// ---------------------------------------------------------------------------
// MultiHeadSelfAttention for MI455X (gfx1250, wave32, WMMA).
// Pipeline (all bf16 intermediates, f32 accumulate in WMMA):
//   1) convert inputs f32->bf16, convert+transpose weights f32->bf16 (Wt[N][K])
//   2) Q/K/V projections: bf16 GEMM + bias -> bf16 (v_wmma_f32_16x16x32_bf16)
//      with double-buffered async (ASYNCcnt) LDS staging of the weight tiles
//   3) flash attention per (b, head, 64-query block), mask + online softmax,
//      async K staging, prefetched mask
//   4) output GEMM + bias -> f32
// ---------------------------------------------------------------------------

typedef __bf16 v16bf __attribute__((ext_vector_type(16)));
typedef __bf16 v8bf  __attribute__((ext_vector_type(8)));
typedef __bf16 v4bf  __attribute__((ext_vector_type(4)));
typedef float  v8f   __attribute__((ext_vector_type(8)));
typedef int    v4i   __attribute__((ext_vector_type(4)));

union Frag16 { v16bf f; v8bf h[2]; };

// ---- async memory->LDS (ASYNCcnt) helpers, probed at compile time ---------
#if __has_builtin(__builtin_amdgcn_global_load_async_to_lds_b128)
#define HAVE_ASYNC_LDS 1
__device__ __forceinline__ void async_copy16(__bf16* l, const __bf16* g) {
    // b128 payload type is int4 per the toolchain's builtin signature
    __builtin_amdgcn_global_load_async_to_lds_b128(
        (v4i*)(g), (v4i*)(l), 0, 0);
}
#else
#define HAVE_ASYNC_LDS 0
__device__ __forceinline__ void async_copy16(__bf16* l, const __bf16* g) {
    *(v8bf*)l = *(const v8bf*)g;     // synchronous fallback (16B)
}
#endif

__device__ __forceinline__ void wait_async0() {
#if __has_builtin(__builtin_amdgcn_s_wait_asynccnt)
    __builtin_amdgcn_s_wait_asynccnt(0);
#else
    asm volatile("s_wait_asynccnt 0x0" ::: "memory");
#endif
}

__device__ __forceinline__ __bf16 f2bf(float f) {
    union { float f; unsigned u; } a; a.f = f;
    unsigned r = a.u + 0x7FFFu + ((a.u >> 16) & 1u);   // round-to-nearest-even
    union { unsigned short s; __bf16 b; } o; o.s = (unsigned short)(r >> 16);
    return o.b;
}

// ------------------------------- converters --------------------------------

__global__ void conv_f32_bf16(const float* __restrict__ in,
                              __bf16* __restrict__ out, int n4) {
    int i = blockIdx.x * blockDim.x + threadIdx.x;
    if (i < n4) {
        float4 f = ((const float4*)in)[i];
        v4bf o;
        o[0] = f2bf(f.x); o[1] = f2bf(f.y); o[2] = f2bf(f.z); o[3] = f2bf(f.w);
        ((v4bf*)out)[i] = o;
    }
}

// W[K=1024][N=1024] f32 -> Wt[N][K] bf16 (so B-fragments read contiguous K)
__global__ void conv_transpose_w(const float* __restrict__ Wsrc,
                                 __bf16* __restrict__ Wt) {
    int i = blockIdx.x * blockDim.x + threadIdx.x;   // 0 .. 1024*1024-1
    int k = i >> 10, n = i & 1023;
    Wt[(size_t)n * 1024 + k] = f2bf(Wsrc[i]);
}

// ------------------------------- bf16 GEMM ---------------------------------
// C[M][N] = A[M][K] @ Wt[N][K]^T + bias, 256 thr = 8 waves, tile 128x128.
// Wave w owns a 16-row strip, 8 n-tiles of 16; K stepped by 32.
// Weight tiles are double-buffered in LDS and filled with async LDS-DMA so
// the next tile streams in while the current tile's 8 WMMAs execute.

template <bool F32OUT>
__global__ void gemm_bias(const __bf16* __restrict__ A,
                          const __bf16* __restrict__ Wt,
                          const float* __restrict__ bias,
                          void* __restrict__ Cout,
                          int M, int N, int K) {
    __shared__ __align__(32) __bf16 ldsB[2][128 * 32];   // [buf][n_local][k_local]

    const int t  = threadIdx.x;
    const int w  = t >> 5;
    const int l  = t & 31;
    const int hi = l >> 4;
    const int li = l & 15;
    const int m0 = blockIdx.y * 128;
    const int n0 = blockIdx.x * 128;

    // staging role: 128 n-rows x 32 k, 32B (two b128) per thread
    const int sn = t >> 1, sh = (t & 1) * 16;
    const __bf16* wrow = Wt + (size_t)(n0 + sn) * K + sh;
    __bf16* sdst0 = &ldsB[0][sn * 32 + sh];
    __bf16* sdst1 = &ldsB[1][sn * 32 + sh];

    v8f acc[8] = {};
    const __bf16* arow = A + (size_t)(m0 + w * 16 + li) * K;

    // prologue: stage k0 = 0 into buffer 0
    async_copy16(sdst0,     wrow);
    async_copy16(sdst0 + 8, wrow + 8);
    wait_async0();
    __syncthreads();

    const int nIter = K / 32;
    for (int it = 0; it < nIter; ++it) {
        const int k0  = it * 32;
        const int cur = it & 1;
        if (it + 1 < nIter) {   // kick off next tile's DMA into the other buffer
            __bf16* nd = (cur ? sdst0 : sdst1);
            async_copy16(nd,     wrow + k0 + 32);
            async_copy16(nd + 8, wrow + k0 + 40);
        }

        Frag16 a;
        a.h[0] = *(const v8bf*)(arow + k0 + hi * 16);
        a.h[1] = *(const v8bf*)(arow + k0 + hi * 16 + 8);

#pragma unroll
        for (int j = 0; j < 8; ++j) {
            v16bf b = *(const v16bf*)&ldsB[cur][(j * 16 + li) * 32 + hi * 16];
            acc[j] = __builtin_amdgcn_wmma_f32_16x16x32_bf16(
                false, a.f, false, b, (short)0, acc[j], false, false);
        }

        wait_async0();          // this wave's DMA into next buffer landed
        __syncthreads();        // everyone's DMA landed, everyone's reads done
    }

#pragma unroll
    for (int j = 0; j < 8; ++j) {
        int col = n0 + j * 16 + li;
        float bv = bias[col];
#pragma unroll
        for (int v = 0; v < 8; ++v) {
            int row = m0 + w * 16 + v + hi * 8;
            float val = acc[j][v] + bv;
            if constexpr (F32OUT)
                ((float*)Cout)[(size_t)row * N + col] = val;
            else
                ((__bf16*)Cout)[(size_t)row * N + col] = f2bf(val);
        }
    }
}

// ---------------------------- flash attention ------------------------------
// grid (S/64, heads, batch); 128 thr = 4 waves; wave owns 16 query rows.
// Streams keys/values in blocks of 32; K staged with async LDS-DMA (row-major),
// V staged transposed so every WMMA fragment read is a contiguous 32B chunk.

#define S_LEN 2048
#define H_DIM 1024

__global__ void attn_kernel(const __bf16* __restrict__ Qp,
                            const __bf16* __restrict__ Kp,
                            const __bf16* __restrict__ Vp,
                            const int* __restrict__ mask,
                            __bf16* __restrict__ ctx) {
    __shared__ __align__(32) __bf16 ldsK[32 * 64];       // [key][d]
    __shared__ __align__(32) __bf16 ldsV[64 * 32];       // [d][key] (transposed)
    __shared__ __align__(32) __bf16 ldsP[4 * 16 * 32];   // per-wave P tiles

    const int t  = threadIdx.x;
    const int w  = t >> 5;
    const int l  = t & 31;
    const int hi = l >> 4;
    const int li = l & 15;
    const int qb = blockIdx.x;
    const int h  = blockIdx.y;
    const int bz = blockIdx.z;
    const int q0 = qb * 64 + w * 16;

    // Q fragments for this wave's 16 rows (d split 0..31 / 32..63), kept in regs
    const __bf16* qbase =
        Qp + ((size_t)bz * S_LEN + q0 + li) * H_DIM + h * 64;
    Frag16 a0, a1;
    a0.h[0] = *(const v8bf*)(qbase + hi * 16);
    a0.h[1] = *(const v8bf*)(qbase + hi * 16 + 8);
    a1.h[0] = *(const v8bf*)(qbase + 32 + hi * 16);
    a1.h[1] = *(const v8bf*)(qbase + 40 + hi * 16);

    v8f acc[4] = {};
    float mrun[8], lrun[8];
#pragma unroll
    for (int v = 0; v < 8; ++v) { mrun[v] = -3.0e38f; lrun[v] = 0.0f; }

    const int r   = t >> 2;          // staging: key row 0..31
    const int c16 = (t & 3) * 16;    // staging: d chunk

    for (int kb = 0; kb < S_LEN / 32; ++kb) {
        // ---- staging: K via async LDS-DMA, V via load + transpose scatter ----
        {
            size_t src = ((size_t)bz * S_LEN + kb * 32 + r) * H_DIM + h * 64 + c16;
            async_copy16(&ldsK[r * 64 + c16],     &Kp[src]);      // overlaps with
            async_copy16(&ldsK[r * 64 + c16 + 8], &Kp[src + 8]);  // the V scatter
            v16bf vv = *(const v16bf*)&Vp[src];
#pragma unroll
            for (int e = 0; e < 16; ++e)
                ldsV[(c16 + e) * 32 + r] = vv[e];
        }
        // prefetch next block's mask rows while DMA/scatter complete
        if (kb + 1 < S_LEN / 32)
            __builtin_prefetch(
                mask + ((size_t)bz * S_LEN + q0 + hi * 8) * S_LEN + (kb + 1) * 32 + li,
                0, 1);
        wait_async0();
        __syncthreads();

        // ---- scores: two 16x16 tiles (32 keys), K-dim 64 = 2 chained WMMAs ----
        v8f st[2];
#pragma unroll
        for (int nt = 0; nt < 2; ++nt) {
            v16bf b0 = *(const v16bf*)&ldsK[(nt * 16 + li) * 64 + hi * 16];
            v16bf b1 = *(const v16bf*)&ldsK[(nt * 16 + li) * 64 + 32 + hi * 16];
            v8f z = {};
            v8f s = __builtin_amdgcn_wmma_f32_16x16x32_bf16(
                false, a0.f, false, b0, (short)0, z, false, false);
            s = __builtin_amdgcn_wmma_f32_16x16x32_bf16(
                false, a1.f, false, b1, (short)0, s, false, false);
            st[nt] = s;
        }

        // ---- scale, mask, online softmax (row = v + hi*8, cols across lanes) --
        float p0[8], p1[8], corr[8];
#pragma unroll
        for (int v = 0; v < 8; ++v) {
            int qr = q0 + v + hi * 8;
            const int* mp = mask + ((size_t)bz * S_LEN + qr) * S_LEN + kb * 32;
            float s0 = st[0][v] * 0.125f; if (mp[li] == 0)      s0 = -1.0e6f;
            float s1 = st[1][v] * 0.125f; if (mp[16 + li] == 0) s1 = -1.0e6f;
            float mx = fmaxf(s0, s1);
#pragma unroll
            for (int o = 1; o < 16; o <<= 1)
                mx = fmaxf(mx, __shfl_xor(mx, o, 32));
            float mnew = fmaxf(mrun[v], mx);
            float c = __expf(mrun[v] - mnew);
            mrun[v] = mnew; corr[v] = c;
            s0 = __expf(s0 - mnew); s1 = __expf(s1 - mnew);
            float sum = s0 + s1;
#pragma unroll
            for (int o = 1; o < 16; o <<= 1)
                sum += __shfl_xor(sum, o, 32);
            lrun[v] = lrun[v] * c + sum;
            p0[v] = s0; p1[v] = s1;
        }
#pragma unroll
        for (int dc = 0; dc < 4; ++dc)
#pragma unroll
            for (int v = 0; v < 8; ++v) acc[dc][v] *= corr[v];

        // ---- re-layout P (C-layout -> A-fragment) via per-wave LDS ----
        __bf16* myP = &ldsP[w * (16 * 32)];
#pragma unroll
        for (int v = 0; v < 8; ++v) {
            int m = v + hi * 8;
            myP[m * 32 + li]      = f2bf(p0[v]);
            myP[m * 32 + 16 + li] = f2bf(p1[v]);
        }
        asm volatile("s_wait_dscnt 0" ::: "memory");   // wave-local RAW on LDS
        Frag16 pa;
        pa.h[0] = *(const v8bf*)&myP[li * 32 + hi * 16];
        pa.h[1] = *(const v8bf*)&myP[li * 32 + hi * 16 + 8];

        // ---- O += P @ V  (4 d-chunks of 16, K-dim 32) ----
#pragma unroll
        for (int dc = 0; dc < 4; ++dc) {
            v16bf vb = *(const v16bf*)&ldsV[(dc * 16 + li) * 32 + hi * 16];
            acc[dc] = __builtin_amdgcn_wmma_f32_16x16x32_bf16(
                false, pa.f, false, vb, (short)0, acc[dc], false, false);
        }
        __syncthreads();
    }

    // ---- normalize and store context (merged-head layout) ----
#pragma unroll
    for (int v = 0; v < 8; ++v) {
        float inv = 1.0f / lrun[v];
        int qr = q0 + v + hi * 8;
        __bf16* cp = ctx + ((size_t)bz * S_LEN + qr) * H_DIM + h * 64;
#pragma unroll
        for (int dc = 0; dc < 4; ++dc)
            cp[dc * 16 + li] = f2bf(acc[dc][v] * inv);
    }
}

// ------------------------------- launcher ----------------------------------

extern "C" void kernel_launch(void* const* d_in, const int* in_sizes, int n_in,
                              void* d_out, int out_size, void* d_ws, size_t ws_size,
                              hipStream_t stream) {
    (void)in_sizes; (void)n_in; (void)out_size; (void)ws_size;

    const float* q_in = (const float*)d_in[0];
    const float* k_in = (const float*)d_in[1];
    const float* v_in = (const float*)d_in[2];
    const int*   msk  = (const int*)d_in[3];
    const float* Wq = (const float*)d_in[4];  const float* bq = (const float*)d_in[5];
    const float* Wk = (const float*)d_in[6];  const float* bk = (const float*)d_in[7];
    const float* Wv = (const float*)d_in[8];  const float* bv = (const float*)d_in[9];
    const float* Wo = (const float*)d_in[10]; const float* bo = (const float*)d_in[11];

    constexpr size_t Bb = 4, Ss = 2048, Hh = 1024;
    constexpr size_t BSH = Bb * Ss * Hh;      // 8,388,608
    constexpr size_t HH  = Hh * Hh;           // 1,048,576

    __bf16* ws  = (__bf16*)d_ws;
    __bf16* Xb  = ws;                 // 3*BSH : bf16 inputs
    __bf16* Wt  = ws + 3 * BSH;       // 4*HH  : transposed bf16 weights
    __bf16* Pr  = Wt + 4 * HH;        // 3*BSH : Q,K,V projections
    __bf16* Ctx = Pr + 3 * BSH;       // BSH   : attention context

    const dim3 cb(256);
    const int n4 = (int)(BSH / 4);

    conv_f32_bf16<<<dim3(n4 / 256), cb, 0, stream>>>(q_in, Xb,            n4);
    conv_f32_bf16<<<dim3(n4 / 256), cb, 0, stream>>>(k_in, Xb + BSH,      n4);
    conv_f32_bf16<<<dim3(n4 / 256), cb, 0, stream>>>(v_in, Xb + 2 * BSH,  n4);

    conv_transpose_w<<<dim3(HH / 256), cb, 0, stream>>>(Wq, Wt);
    conv_transpose_w<<<dim3(HH / 256), cb, 0, stream>>>(Wk, Wt + HH);
    conv_transpose_w<<<dim3(HH / 256), cb, 0, stream>>>(Wv, Wt + 2 * HH);
    conv_transpose_w<<<dim3(HH / 256), cb, 0, stream>>>(Wo, Wt + 3 * HH);

    const dim3 gg(1024 / 128, 8192 / 128);    // (N tiles, M tiles)
    gemm_bias<false><<<gg, cb, 0, stream>>>(Xb,           Wt,          bq, Pr,           8192, 1024, 1024);
    gemm_bias<false><<<gg, cb, 0, stream>>>(Xb + BSH,     Wt + HH,     bk, Pr + BSH,     8192, 1024, 1024);
    gemm_bias<false><<<gg, cb, 0, stream>>>(Xb + 2 * BSH, Wt + 2 * HH, bv, Pr + 2 * BSH, 8192, 1024, 1024);

    attn_kernel<<<dim3(Ss / 64, 16, Bb), dim3(128), 0, stream>>>(
        Pr, Pr + BSH, Pr + 2 * BSH, msk, Ctx);

    gemm_bias<true><<<gg, cb, 0, stream>>>(Ctx, Wt + 3 * HH, bo, d_out, 8192, 1024, 1024);
}
